// HeteroGNN_20383914787255
// MI455X (gfx1250) — compile-verified, matrix operand order checked by reading, not run on hardware
//
#include <hip/hip_runtime.h>
#include <hip/hip_bf16.h>

typedef _Float16 v16h __attribute__((ext_vector_type(16)));
typedef _Float16 v8h  __attribute__((ext_vector_type(8)));
typedef _Float16 v4h  __attribute__((ext_vector_type(4)));
typedef float    v8f  __attribute__((ext_vector_type(8)));

union Frag16 { v16h v; v8h h[2]; };

static constexpr int N_P   = 200000;
static constexpr int N_A   = 150000;
static constexpr long long E_W = 4000000;
static constexpr long long E_R = 4000000;
static constexpr long long E_C = 6000000;
static constexpr int C_IN  = 128;
static constexpr int C_HID = 256;
static constexpr int C_OUT = 349;
static constexpr int C_OUT_PAD = 384;   // 6 strips of 64
static constexpr int K_MAX = 256;
static constexpr int LDS_PAD = 8;       // halves; 16B row-stride pad -> conflict-free b128 reads

// ---------------------------------------------------------------- utilities
__global__ void k_fill_f32(float* __restrict__ p, unsigned n, float v) {
    unsigned i = blockIdx.x * blockDim.x + threadIdx.x;
    if (i < n) p[i] = v;
}

__global__ void k_cvt_f16(const float* __restrict__ in, _Float16* __restrict__ out, unsigned n) {
    unsigned i = blockIdx.x * blockDim.x + threadIdx.x;
    if (i < n) out[i] = (_Float16)in[i];
}

__global__ void k_relu_cvt_f16(const float* __restrict__ in, _Float16* __restrict__ out, unsigned n) {
    unsigned i = blockIdx.x * blockDim.x + threadIdx.x;
    if (i < n) out[i] = (_Float16)fmaxf(in[i], 0.0f);
}

// W: [K x N] f32 row-major  ->  Wt: [Npad x K] f16 row-major (transposed, zero padded)
__global__ void k_cvt_weight_t(const float* __restrict__ W, _Float16* __restrict__ Wt,
                               int K, int N, unsigned total) {
    unsigned i = blockIdx.x * blockDim.x + threadIdx.x;
    if (i >= total) return;
    unsigned n = i / (unsigned)K;
    unsigned k = i - n * (unsigned)K;
    Wt[i] = (n < (unsigned)N) ? (_Float16)W[(size_t)k * N + n] : (_Float16)0.0f;
}

// ----------------------------------------------------------- edge degree
__global__ void k_count(const int* __restrict__ ei, unsigned E, float* __restrict__ cnt) {
    unsigned i = blockIdx.x * blockDim.x + threadIdx.x;
    if (i < E) atomicAdd(&cnt[ei[(size_t)E + i]], 1.0f);
}

// ----------------------------------------------------------- scatter-add (f32 source)
// one thread handles 4 channels of one edge; float4 gather, 4 f32 atomics
__global__ void k_scatter_f32(const float* __restrict__ x, const int* __restrict__ ei,
                              unsigned E, int C, float* __restrict__ out) {
    unsigned i  = blockIdx.x * blockDim.x + threadIdx.x;
    unsigned cq = (unsigned)(C >> 2);
    unsigned total = E * cq;
    if (i >= total) return;
    unsigned e = i / cq;
    unsigned q = i - e * cq;
    int s = ei[e];
    int d = ei[(size_t)E + e];
    const float4 v = ((const float4*)(x + (size_t)s * C))[q];
    float* o = out + (size_t)d * C + q * 4u;
    atomicAdd(o + 0, v.x); atomicAdd(o + 1, v.y);
    atomicAdd(o + 2, v.z); atomicAdd(o + 3, v.w);
}

// ----------------------------------------------------------- scatter-add (f16 source)
__global__ void k_scatter_f16(const _Float16* __restrict__ x, const int* __restrict__ ei,
                              unsigned E, int C, float* __restrict__ out) {
    unsigned i  = blockIdx.x * blockDim.x + threadIdx.x;
    unsigned cq = (unsigned)(C >> 2);
    unsigned total = E * cq;
    if (i >= total) return;
    unsigned e = i / cq;
    unsigned q = i - e * cq;
    int s = ei[e];
    int d = ei[(size_t)E + e];
    const v4h v = ((const v4h*)(x + (size_t)s * C))[q];
    float* o = out + (size_t)d * C + q * 4u;
    atomicAdd(o + 0, (float)v.x); atomicAdd(o + 1, (float)v.y);
    atomicAdd(o + 2, (float)v.z); atomicAdd(o + 3, (float)v.w);
}

// mean-normalize (divide by max(deg,1)) and convert to f16 for WMMA input
__global__ void k_norm_f16(const float* __restrict__ agg, const float* __restrict__ cnt,
                           _Float16* __restrict__ out, unsigned n, int C) {
    unsigned i = blockIdx.x * blockDim.x + threadIdx.x;
    if (i >= n) return;
    unsigned node = i / (unsigned)C;
    out[i] = (_Float16)(agg[i] / fmaxf(cnt[node], 1.0f));
}

// ----------------------------------------------------------- WMMA GEMM
// C[M x ldc](f32) (+)= A[M x K](f16 row-major) * Bt[Npad x K](f16, B transposed) + bias
// block = 256 threads = 8 waves; wave -> 16(M) x 64(N) tile via 4 accumulators.
// The 64-col B strip (shared by all 8 waves) is staged once into LDS with a padded
// row stride so the 16-lane ds_load_b128 fragment reads are bank-conflict free.
__global__ __launch_bounds__(256)
void k_gemm_wmma(const _Float16* __restrict__ A, const _Float16* __restrict__ Bt,
                 float* __restrict__ C, const float* __restrict__ bias,
                 int M, int Nstore, int ldc, int K, int accumulate) {
    __shared__ __align__(16) _Float16 sB[64 * (K_MAX + LDS_PAD)];
    const int LDK = K + LDS_PAD;

    const int wave = threadIdx.x >> 5;
    const int lane = threadIdx.x & 31;
    const int n0   = blockIdx.y * 64;

    // ---- cooperative stage of B strip [64 cols x K] into LDS (all waves participate)
    {
        const unsigned chunksPerCol = (unsigned)K >> 3;        // 8 halves (16B) per chunk
        const unsigned nChunks = 64u * chunksPerCol;
        for (unsigned idx = threadIdx.x; idx < nChunks; idx += 256u) {
            unsigned col = idx / chunksPerCol;
            unsigned ho  = (idx - col * chunksPerCol) * 8u;
            *(v8h*)(sB + col * (unsigned)LDK + ho) =
                *(const v8h*)(Bt + (size_t)(n0 + col) * K + ho);
        }
    }
    __syncthreads();

    const int m0 = (blockIdx.x * 8 + wave) * 16;
    if (m0 >= M) return;                       // wave-uniform: EXEC stays all-ones
    const int row  = lane & 15;
    const int hi   = lane >> 4;                // K half-select per ISA A/B layout
    const int koff = hi * 8;

    const _Float16* Arow = A + (size_t)(m0 + row) * K;

    v8f acc[4] = {};
    for (int k = 0; k < K; k += 32) {
        Frag16 a;
        a.h[0] = *(const v8h*)(Arow + k + koff);        // K = koff..koff+7
        a.h[1] = *(const v8h*)(Arow + k + koff + 16);   // K = koff+16..koff+23
#pragma unroll
        for (int j = 0; j < 4; ++j) {
            const _Float16* Bcol = sB + (unsigned)(j * 16 + row) * (unsigned)LDK;
            Frag16 b;
            b.h[0] = *(const v8h*)(Bcol + k + koff);
            b.h[1] = *(const v8h*)(Bcol + k + koff + 16);
            acc[j] = __builtin_amdgcn_wmma_f32_16x16x32_f16(
                false, a.v, false, b.v, (short)0, acc[j], false, false);
        }
    }

#pragma unroll
    for (int j = 0; j < 4; ++j) {
        int n = n0 + j * 16 + row;
        if (n >= Nstore) continue;
        float bv = bias ? bias[n] : 0.0f;
#pragma unroll
        for (int r = 0; r < 8; ++r) {
            size_t idx = (size_t)(m0 + hi * 8 + r) * ldc + n;
            float v = acc[j][r] + bv;
            C[idx] = accumulate ? (C[idx] + v) : v;
        }
    }
}

// ---------------------------------------------------------------- launcher
extern "C" void kernel_launch(void* const* d_in, const int* in_sizes, int n_in,
                              void* d_out, int out_size, void* d_ws, size_t ws_size,
                              hipStream_t stream) {
    const float* x_paper  = (const float*)d_in[0];
    const float* x_author = (const float*)d_in[1];
    const int*   ei_w = (const int*)d_in[2];
    const int*   ei_r = (const int*)d_in[3];
    const int*   ei_c = (const int*)d_in[4];
    const float* c1_w_Wl = (const float*)d_in[5];
    const float* c1_w_Wr = (const float*)d_in[6];
    const float* c1_w_b  = (const float*)d_in[7];
    const float* c2_w_Wl = (const float*)d_in[8];
    const float* c2_w_Wr = (const float*)d_in[9];
    const float* c2_w_b  = (const float*)d_in[10];
    const float* c1_r_Wl = (const float*)d_in[11];
    const float* c1_r_Wr = (const float*)d_in[12];
    const float* c1_r_b  = (const float*)d_in[13];
    const float* c2_r_Wl = (const float*)d_in[14];
    const float* c2_r_Wr = (const float*)d_in[15];
    const float* c2_r_b  = (const float*)d_in[16];
    const float* c1_c_Wl = (const float*)d_in[17];
    const float* c1_c_Wr = (const float*)d_in[18];
    const float* c1_c_b  = (const float*)d_in[19];
    const float* c2_c_Wl = (const float*)d_in[20];
    const float* c2_c_Wr = (const float*)d_in[21];
    const float* c2_c_b  = (const float*)d_in[22];
    const float* lin_p_W = (const float*)d_in[23];
    const float* lin_p_b = (const float*)d_in[24];
    const float* lin_a_W = (const float*)d_in[25];
    const float* lin_a_b = (const float*)d_in[26];
    float* out = (float*)d_out;

    // ---- workspace layout (256B aligned slices)
    char*  ws  = (char*)d_ws;
    size_t off = 0;
    auto alloc = [&](size_t bytes) -> char* {
        char* p = ws + off;
        off = (off + bytes + 255) & ~(size_t)255;
        return p;
    };
    float*    cnt_w   = (float*)alloc((size_t)N_P * 4);
    float*    cnt_c   = (float*)alloc((size_t)N_P * 4);
    float*    cnt_r   = (float*)alloc((size_t)N_A * 4);
    float*    tmp_agg = (float*)alloc((size_t)N_P * C_HID * 4);   // f32 scatter target
    _Float16* agg_h   = (_Float16*)alloc((size_t)N_P * C_HID * 2);// normalized f16 agg
    float*    acc     = (float*)alloc((size_t)N_P * C_HID * 4);   // pre-activation f32
    _Float16* xp_h    = (_Float16*)alloc((size_t)N_P * C_IN * 2);
    _Float16* xa_h    = (_Float16*)alloc((size_t)N_A * C_IN * 2);
    _Float16* hp_h    = (_Float16*)alloc((size_t)N_P * C_HID * 2);
    _Float16* ha_h    = (_Float16*)alloc((size_t)N_A * C_HID * 2);
    _Float16* gp_h    = (_Float16*)alloc((size_t)N_P * C_HID * 2);
    _Float16* ga_h    = (_Float16*)alloc((size_t)N_A * C_HID * 2);
    auto wt1 = [&]() { return (_Float16*)alloc((size_t)C_HID * C_IN * 2); };   // 256x128
    auto wt2 = [&]() { return (_Float16*)alloc((size_t)C_HID * C_HID * 2); };  // 256x256
    _Float16* t_c1_w_Wl = wt1(); _Float16* t_c1_w_Wr = wt1();
    _Float16* t_c1_r_Wl = wt1(); _Float16* t_c1_r_Wr = wt1();
    _Float16* t_c1_c_Wl = wt1(); _Float16* t_c1_c_Wr = wt1();
    _Float16* t_c2_w_Wl = wt2(); _Float16* t_c2_w_Wr = wt2();
    _Float16* t_c2_r_Wl = wt2(); _Float16* t_c2_r_Wr = wt2();
    _Float16* t_c2_c_Wl = wt2(); _Float16* t_c2_c_Wr = wt2();
    _Float16* t_lin_p = (_Float16*)alloc((size_t)C_OUT_PAD * C_HID * 2);       // 384x256
    _Float16* t_lin_a = (_Float16*)alloc((size_t)C_OUT_PAD * C_HID * 2);

    auto g1 = [](unsigned long long total) { return dim3((unsigned)((total + 255) / 256)); };
    dim3 B(256);

    auto fill = [&](float* p, unsigned n) {
        k_fill_f32<<<g1(n), B, 0, stream>>>(p, n, 0.0f);
    };
    auto cvt_w = [&](const float* W, _Float16* Wt, int K, int N, int Npad) {
        unsigned total = (unsigned)Npad * (unsigned)K;
        k_cvt_weight_t<<<g1(total), B, 0, stream>>>(W, Wt, K, N, total);
    };
    auto gemm = [&](const _Float16* A, const _Float16* Bt, float* Cp, const float* bias,
                    int M, int Nstore, int ldc, int K, int Npad, int accum) {
        dim3 grid((unsigned)((M + 127) / 128), (unsigned)(Npad / 64));
        k_gemm_wmma<<<grid, B, 0, stream>>>(A, Bt, Cp, bias, M, Nstore, ldc, K, accum);
    };

    // ================= prep: f16 feature copies, transposed f16 weights, degrees
    k_cvt_f16<<<g1((size_t)N_P * C_IN), B, 0, stream>>>(x_paper,  xp_h, (unsigned)(N_P * C_IN));
    k_cvt_f16<<<g1((size_t)N_A * C_IN), B, 0, stream>>>(x_author, xa_h, (unsigned)(N_A * C_IN));
    cvt_w(c1_w_Wl, t_c1_w_Wl, C_IN,  C_HID, C_HID); cvt_w(c1_w_Wr, t_c1_w_Wr, C_IN,  C_HID, C_HID);
    cvt_w(c1_r_Wl, t_c1_r_Wl, C_IN,  C_HID, C_HID); cvt_w(c1_r_Wr, t_c1_r_Wr, C_IN,  C_HID, C_HID);
    cvt_w(c1_c_Wl, t_c1_c_Wl, C_IN,  C_HID, C_HID); cvt_w(c1_c_Wr, t_c1_c_Wr, C_IN,  C_HID, C_HID);
    cvt_w(c2_w_Wl, t_c2_w_Wl, C_HID, C_HID, C_HID); cvt_w(c2_w_Wr, t_c2_w_Wr, C_HID, C_HID, C_HID);
    cvt_w(c2_r_Wl, t_c2_r_Wl, C_HID, C_HID, C_HID); cvt_w(c2_r_Wr, t_c2_r_Wr, C_HID, C_HID, C_HID);
    cvt_w(c2_c_Wl, t_c2_c_Wl, C_HID, C_HID, C_HID); cvt_w(c2_c_Wr, t_c2_c_Wr, C_HID, C_HID, C_HID);
    cvt_w(lin_p_W, t_lin_p, C_HID, C_OUT, C_OUT_PAD);
    cvt_w(lin_a_W, t_lin_a, C_HID, C_OUT, C_OUT_PAD);

    fill(cnt_w, N_P); fill(cnt_c, N_P); fill(cnt_r, N_A);
    k_count<<<g1(E_W), B, 0, stream>>>(ei_w, (unsigned)E_W, cnt_w);
    k_count<<<g1(E_C), B, 0, stream>>>(ei_c, (unsigned)E_C, cnt_c);
    k_count<<<g1(E_R), B, 0, stream>>>(ei_r, (unsigned)E_R, cnt_r);

    // ================= layer 1, paper:  relu( SAGE(writes) + SAGE(cites) )
    fill(acc, (unsigned)(N_P * C_HID));
    // writes: author -> paper
    fill(tmp_agg, (unsigned)(N_P * C_IN));
    k_scatter_f32<<<g1(E_W * (C_IN / 4)), B, 0, stream>>>(x_author, ei_w, (unsigned)E_W, C_IN, tmp_agg);
    k_norm_f16<<<g1((size_t)N_P * C_IN), B, 0, stream>>>(tmp_agg, cnt_w, agg_h, (unsigned)(N_P * C_IN), C_IN);
    gemm(agg_h, t_c1_w_Wl, acc, c1_w_b, N_P, C_HID, C_HID, C_IN, C_HID, 1);
    gemm(xp_h,  t_c1_w_Wr, acc, nullptr, N_P, C_HID, C_HID, C_IN, C_HID, 1);
    // cites: paper -> paper
    fill(tmp_agg, (unsigned)(N_P * C_IN));
    k_scatter_f32<<<g1(E_C * (C_IN / 4)), B, 0, stream>>>(x_paper, ei_c, (unsigned)E_C, C_IN, tmp_agg);
    k_norm_f16<<<g1((size_t)N_P * C_IN), B, 0, stream>>>(tmp_agg, cnt_c, agg_h, (unsigned)(N_P * C_IN), C_IN);
    gemm(agg_h, t_c1_c_Wl, acc, c1_c_b, N_P, C_HID, C_HID, C_IN, C_HID, 1);
    gemm(xp_h,  t_c1_c_Wr, acc, nullptr, N_P, C_HID, C_HID, C_IN, C_HID, 1);
    k_relu_cvt_f16<<<g1((size_t)N_P * C_HID), B, 0, stream>>>(acc, hp_h, (unsigned)(N_P * C_HID));

    // ================= layer 1, author: relu( SAGE(rev_writes) )
    fill(acc, (unsigned)(N_A * C_HID));
    fill(tmp_agg, (unsigned)(N_A * C_IN));
    k_scatter_f32<<<g1(E_R * (C_IN / 4)), B, 0, stream>>>(x_paper, ei_r, (unsigned)E_R, C_IN, tmp_agg);
    k_norm_f16<<<g1((size_t)N_A * C_IN), B, 0, stream>>>(tmp_agg, cnt_r, agg_h, (unsigned)(N_A * C_IN), C_IN);
    gemm(agg_h, t_c1_r_Wl, acc, c1_r_b, N_A, C_HID, C_HID, C_IN, C_HID, 1);
    gemm(xa_h,  t_c1_r_Wr, acc, nullptr, N_A, C_HID, C_HID, C_IN, C_HID, 1);
    k_relu_cvt_f16<<<g1((size_t)N_A * C_HID), B, 0, stream>>>(acc, ha_h, (unsigned)(N_A * C_HID));

    // ================= layer 2, paper
    fill(acc, (unsigned)(N_P * C_HID));
    fill(tmp_agg, (unsigned)(N_P * C_HID));
    k_scatter_f16<<<g1(E_W * (C_HID / 4)), B, 0, stream>>>(ha_h, ei_w, (unsigned)E_W, C_HID, tmp_agg);
    k_norm_f16<<<g1((size_t)N_P * C_HID), B, 0, stream>>>(tmp_agg, cnt_w, agg_h, (unsigned)(N_P * C_HID), C_HID);
    gemm(agg_h, t_c2_w_Wl, acc, c2_w_b, N_P, C_HID, C_HID, C_HID, C_HID, 1);
    gemm(hp_h,  t_c2_w_Wr, acc, nullptr, N_P, C_HID, C_HID, C_HID, C_HID, 1);
    fill(tmp_agg, (unsigned)(N_P * C_HID));
    k_scatter_f16<<<g1(E_C * (C_HID / 4)), B, 0, stream>>>(hp_h, ei_c, (unsigned)E_C, C_HID, tmp_agg);
    k_norm_f16<<<g1((size_t)N_P * C_HID), B, 0, stream>>>(tmp_agg, cnt_c, agg_h, (unsigned)(N_P * C_HID), C_HID);
    gemm(agg_h, t_c2_c_Wl, acc, c2_c_b, N_P, C_HID, C_HID, C_HID, C_HID, 1);
    gemm(hp_h,  t_c2_c_Wr, acc, nullptr, N_P, C_HID, C_HID, C_HID, C_HID, 1);
    k_relu_cvt_f16<<<g1((size_t)N_P * C_HID), B, 0, stream>>>(acc, gp_h, (unsigned)(N_P * C_HID));

    // ================= layer 2, author
    fill(acc, (unsigned)(N_A * C_HID));
    fill(tmp_agg, (unsigned)(N_A * C_HID));
    k_scatter_f16<<<g1(E_R * (C_HID / 4)), B, 0, stream>>>(hp_h, ei_r, (unsigned)E_R, C_HID, tmp_agg);
    k_norm_f16<<<g1((size_t)N_A * C_HID), B, 0, stream>>>(tmp_agg, cnt_r, agg_h, (unsigned)(N_A * C_HID), C_HID);
    gemm(agg_h, t_c2_r_Wl, acc, c2_r_b, N_A, C_HID, C_HID, C_HID, C_HID, 1);
    gemm(ha_h,  t_c2_r_Wr, acc, nullptr, N_A, C_HID, C_HID, C_HID, C_HID, 1);
    k_relu_cvt_f16<<<g1((size_t)N_A * C_HID), B, 0, stream>>>(acc, ga_h, (unsigned)(N_A * C_HID));

    // ================= final per-node-type linear -> d_out (f32, overwrite)
    gemm(gp_h, t_lin_p, out,                         lin_p_b, N_P, C_OUT, C_OUT, C_HID, C_OUT_PAD, 0);
    gemm(ga_h, t_lin_a, out + (size_t)N_P * C_OUT,   lin_a_b, N_A, C_OUT, C_OUT, C_HID, C_OUT_PAD, 0);
}
